// Net_72541997629472
// MI455X (gfx1250) — matrix operand order repaired and checked by reading, not compiled, not run on hardware
//
#include <hip/hip_runtime.h>

// ---------------------------------------------------------------------------
// Problem sizes (fixed by the reference)
// ---------------------------------------------------------------------------
#define N_NODES 524288
#define N_EDGES 2097152
#define N_SUB   8192
#define N_GRP   64
#define F_IN    64
#define HID     128

// ---------------------------------------------------------------------------
// WMMA types / helpers (CDNA5 wave32, 16x16x32 bf16 -> f32)
// ---------------------------------------------------------------------------
typedef __attribute__((ext_vector_type(16))) __bf16 v16bf;
typedef __attribute__((ext_vector_type(8)))  float  v8f;

__device__ __forceinline__ v8f wmma_bf16(v16bf a, v16bf b, v8f c) {
    // (neg_a, A, neg_b, B, c_mod, C, reuse_a, reuse_b)
    return __builtin_amdgcn_wmma_f32_16x16x32_bf16(false, a, false, b,
                                                   (short)0, c, false, false);
}

__device__ __forceinline__ void gatomic_add(float* p, float v) {
    (void)__hip_atomic_fetch_add(p, v, __ATOMIC_RELAXED, __HIP_MEMORY_SCOPE_AGENT);
}
__device__ __forceinline__ void satomic_add(float* p, float v) {
    (void)__hip_atomic_fetch_add(p, v, __ATOMIC_RELAXED, __HIP_MEMORY_SCOPE_WORKGROUP);
}

// A-matrix fragment (16x32 bf16) loaded from an fp32 row-major tile.
// Per ISA 7.12.2: lane holds row M = lane&15; K elements:
//   i in [0,7]  -> K = kbase + i
//   i in [8,15] -> K = kbase + 16 + (i-8)
// where kbase = chunk*32 + (lane>=16 ? 8 : 0).  All four 4-float groups are
// contiguous & 16B-aligned -> 4x global_load_b128 + v_cvt_pk_bf16_f32 packs.
__device__ __forceinline__ v16bf load_a_frag(const float* __restrict__ rowp,
                                             int kbase, float scale) {
    const float4 p0 = *reinterpret_cast<const float4*>(rowp + kbase);
    const float4 p1 = *reinterpret_cast<const float4*>(rowp + kbase + 4);
    const float4 p2 = *reinterpret_cast<const float4*>(rowp + kbase + 16);
    const float4 p3 = *reinterpret_cast<const float4*>(rowp + kbase + 20);
    v16bf a;
    a[0]  = (__bf16)(p0.x * scale); a[1]  = (__bf16)(p0.y * scale);
    a[2]  = (__bf16)(p0.z * scale); a[3]  = (__bf16)(p0.w * scale);
    a[4]  = (__bf16)(p1.x * scale); a[5]  = (__bf16)(p1.y * scale);
    a[6]  = (__bf16)(p1.z * scale); a[7]  = (__bf16)(p1.w * scale);
    a[8]  = (__bf16)(p2.x * scale); a[9]  = (__bf16)(p2.y * scale);
    a[10] = (__bf16)(p2.z * scale); a[11] = (__bf16)(p2.w * scale);
    a[12] = (__bf16)(p3.x * scale); a[13] = (__bf16)(p3.y * scale);
    a[14] = (__bf16)(p3.z * scale); a[15] = (__bf16)(p3.w * scale);
    return a;
}

// B fragment stored fragment-major in LDS: one aligned 32B vector read
// (lowers to two ds_load_b128).
__device__ __forceinline__ v16bf lds_frag(const __bf16* p) {
    return *reinterpret_cast<const v16bf*>(p);
}

// ---------------------------------------------------------------------------
// Kernel 0: zero the accumulator workspace (msg_sum, deg, g, energy, norm)
// ---------------------------------------------------------------------------
__global__ void zero_kernel(float* __restrict__ p, size_t n) {
    size_t i = (size_t)blockIdx.x * blockDim.x + threadIdx.x;
    size_t stride = (size_t)gridDim.x * blockDim.x;
    for (; i < n; i += stride) p[i] = 0.0f;
}

// ---------------------------------------------------------------------------
// Kernel 1: edge scatter (the roofline bottleneck: ~1.1 GB + 134M atomics).
// Each wave owns 32 edges: lanes make ONE coalesced index load each, prefetch
// their own edge's source row (random 256B gather -> global_prefetch_b8 warms
// L2 across the 32-iteration broadcast loop), then the wave walks the edges
// via __shfl broadcast with each lane handling 2 features (coalesced float2
// gather + 2x native global_atomic_add_f32 into msg[dst]).
// ---------------------------------------------------------------------------
__global__ void __launch_bounds__(256)
edge_scatter_kernel(const float* __restrict__ x,
                    const long long* __restrict__ edge_index,
                    float* __restrict__ msg,
                    float* __restrict__ deg) {
    const int wave = threadIdx.x >> 5;
    const int lane = threadIdx.x & 31;
    const int e0   = (blockIdx.x * 8 + wave) * 32;   // N_EDGES % 256 == 0

    const long long s_l = edge_index[e0 + lane];                    // src
    const long long d_l = edge_index[(size_t)N_EDGES + e0 + lane];  // dst

    // Warm L2 with this lane's gather target (row = 256B).
    __builtin_prefetch(x + (size_t)s_l * F_IN, 0, 1);

    // One degree bump per lane-owned edge.
    gatomic_add(deg + d_l, 1.0f);

    const int f = lane * 2;
#pragma unroll 4
    for (int i = 0; i < 32; ++i) {
        const long long s = __shfl(s_l, i);
        const long long d = __shfl(d_l, i);
        const float2 v = *reinterpret_cast<const float2*>(x + (size_t)s * F_IN + f);
        float* mdst = msg + (size_t)d * F_IN + f;
        gatomic_add(mdst + 0, v.x);
        gatomic_add(mdst + 1, v.y);
    }
}

// ---------------------------------------------------------------------------
// Kernel 2: fused SAGE conv + global_add_pool.
//   h = relu((msg/deg) @ W_l + x @ W_r + b_conv)   [16-row tile per wave]
//   g[node_batch[row]] += h[row]                   (atomics into 4MB L2-resident g)
// 8 waves/block -> 128 rows/block.  W_l/W_r staged in LDS, bf16, fragment-major.
// ---------------------------------------------------------------------------
__global__ void __launch_bounds__(256)
conv_pool_kernel(const float* __restrict__ msg, const float* __restrict__ deg,
                 const float* __restrict__ x,
                 const float* __restrict__ W_l, const float* __restrict__ W_r,
                 const float* __restrict__ b_conv,
                 const long long* __restrict__ node_batch,
                 float* __restrict__ g) {
    // [ntile(8)][kchunk(2)][lane(32)][elem(16)] bf16, 16KB each
    __shared__ __align__(32) __bf16 lwl[8 * 2 * 32 * 16];
    __shared__ __align__(32) __bf16 lwr[8 * 2 * 32 * 16];

    const int tid = threadIdx.x;
    // Stage weights: B-fragment layout (ISA 7.12.2): lane holds col N=lane&15,
    // K = kchunk*32 + i + (lane>=16 ? 8 : 0) + (i>=8 ? 8 : 0).
    for (int t = tid; t < 8192; t += 256) {
        const int i  = t & 15;
        const int ln = (t >> 4) & 31;
        const int kc = (t >> 9) & 1;
        const int nt = t >> 10;
        const int K  = kc * 32 + i + ((ln & 16) ? 8 : 0) + ((i & 8) ? 8 : 0);
        const int Nc = nt * 16 + (ln & 15);
        lwl[t] = (__bf16)W_l[K * HID + Nc];
        lwr[t] = (__bf16)W_r[K * HID + Nc];
    }
    __syncthreads();

    const int wave = tid >> 5;
    const int lane = tid & 31;
    const int row0 = (blockIdx.x * 8 + wave) * 16;
    const int m    = lane & 15;
    const int hi   = (lane >> 4) & 1;
    const int row  = row0 + m;

    const float scale = 1.0f / fmaxf(deg[row], 1.0f);

    v16bf a_agg[2], a_x[2];
#pragma unroll
    for (int c = 0; c < 2; ++c) {
        const int kbase = c * 32 + hi * 8;
        a_agg[c] = load_a_frag(msg + (size_t)row * F_IN, kbase, scale);
        a_x[c]   = load_a_frag(x   + (size_t)row * F_IN, kbase, 1.0f);
    }

    // Output rows this lane owns in the D fragment: M = j + hi*8.
    // 32-bit pooled offsets (g is 4MB) -> SADDR+voffset atomic form.
    unsigned segoff[8];
#pragma unroll
    for (int j = 0; j < 8; ++j)
        segoff[j] = (unsigned)node_batch[row0 + j + hi * 8] * HID;

#pragma unroll
    for (int nt = 0; nt < 8; ++nt) {
        const int n = nt * 16 + m;
        const float bias = b_conv[n];
        v8f acc = {bias, bias, bias, bias, bias, bias, bias, bias};
        const __bf16* fl = lwl + ((nt * 2 + 0) * 32 + lane) * 16;
        const __bf16* fr = lwr + ((nt * 2 + 0) * 32 + lane) * 16;
        acc = wmma_bf16(a_agg[0], lds_frag(fl),           acc);
        acc = wmma_bf16(a_agg[1], lds_frag(fl + 32 * 16), acc);
        acc = wmma_bf16(a_x[0],   lds_frag(fr),           acc);
        acc = wmma_bf16(a_x[1],   lds_frag(fr + 32 * 16), acc);
#pragma unroll
        for (int j = 0; j < 8; ++j) {
            const float t = acc[j];
            const float v = (t > 0.0f) ? t : 0.0f;          // relu (select)
            gatomic_add(g + segoff[j] + n, v);              // fused pool
        }
    }
}

// ---------------------------------------------------------------------------
// Kernel 3: MLP head over subgraph embeddings.
//   s = leaky_relu(g@W1 + b1) @ W2 + b2 ;  energy/norm scatter by group.
// 16-row tile per wave, K=128 -> 4 WMMAs per 16x16 tile.
// ---------------------------------------------------------------------------
__global__ void __launch_bounds__(256)
mlp_head_kernel(const float* __restrict__ g,
                const float* __restrict__ W1, const float* __restrict__ b1,
                const float* __restrict__ W2, const float* __restrict__ b2,
                const float* __restrict__ weights,
                const long long* __restrict__ subgraph_batch,
                float* __restrict__ energy, float* __restrict__ norm) {
    // [ntile(8)][kchunk(4)][lane(32)][elem(16)] bf16 = 32KB
    __shared__ __align__(32) __bf16 lw1[8 * 4 * 32 * 16];
    __shared__ float rowsum[8][16];

    const int tid = threadIdx.x;
    for (int t = tid; t < 16384; t += 256) {
        const int i  = t & 15;
        const int ln = (t >> 4) & 31;
        const int kc = (t >> 9) & 3;
        const int nt = t >> 11;
        const int K  = kc * 32 + i + ((ln & 16) ? 8 : 0) + ((i & 8) ? 8 : 0);
        const int Nc = nt * 16 + (ln & 15);
        lw1[t] = (__bf16)W1[K * HID + Nc];
    }
    const int wave = tid >> 5;
    const int lane = tid & 31;
    if (lane < 16) rowsum[wave][lane] = 0.0f;
    __syncthreads();

    const int row0 = (blockIdx.x * 8 + wave) * 16;
    const int m    = lane & 15;
    const int hi   = (lane >> 4) & 1;
    const int row  = row0 + m;

    v16bf a[4];
#pragma unroll
    for (int c = 0; c < 4; ++c)
        a[c] = load_a_frag(g + (size_t)row * HID, c * 32 + hi * 8, 1.0f);

#pragma unroll
    for (int nt = 0; nt < 8; ++nt) {
        const int n = nt * 16 + m;
        const float bias = b1[n];
        v8f acc = {bias, bias, bias, bias, bias, bias, bias, bias};
        const __bf16* f = lw1 + ((nt * 4) * 32 + lane) * 16;
#pragma unroll
        for (int c = 0; c < 4; ++c)
            acc = wmma_bf16(a[c], lds_frag(f + c * 32 * 16), acc);
        const float w2 = W2[n];   // W2 is [HID,1]
#pragma unroll
        for (int j = 0; j < 8; ++j) {
            float v = acc[j];
            v = (v > 0.0f) ? v : 0.01f * v;                 // leaky_relu
            satomic_add(&rowsum[wave][j + hi * 8], v * w2); // ds_add_f32
        }
    }
    __syncthreads();

    if (lane < 16) {
        const int r = row0 + lane;
        const float s  = rowsum[wave][lane] + b2[0];
        const float w  = weights[r];
        const unsigned gb = (unsigned)subgraph_batch[r];
        gatomic_add(energy + gb, s * w);
        gatomic_add(norm + gb, w);
    }
}

// ---------------------------------------------------------------------------
// Kernel 4: final divide
// ---------------------------------------------------------------------------
__global__ void finalize_kernel(const float* __restrict__ energy,
                                const float* __restrict__ norm,
                                float* __restrict__ out) {
    const int i = threadIdx.x;
    if (i < N_GRP) out[i] = energy[i] / norm[i];
}

// ---------------------------------------------------------------------------
// Launch
// ---------------------------------------------------------------------------
extern "C" void kernel_launch(void* const* d_in, const int* in_sizes, int n_in,
                              void* d_out, int out_size, void* d_ws, size_t ws_size,
                              hipStream_t stream) {
    const float*     x              = (const float*)d_in[0];
    const long long* edge_index     = (const long long*)d_in[1];
    const long long* node_batch     = (const long long*)d_in[2];
    const float*     weights        = (const float*)d_in[3];
    const long long* subgraph_batch = (const long long*)d_in[4];
    const float*     W_l            = (const float*)d_in[5];
    const float*     W_r            = (const float*)d_in[6];
    const float*     b_conv         = (const float*)d_in[7];
    const float*     W1             = (const float*)d_in[8];
    const float*     b1             = (const float*)d_in[9];
    const float*     W2             = (const float*)d_in[10];
    const float*     b2             = (const float*)d_in[11];
    float* out = (float*)d_out;

    // Workspace layout (floats)
    float* ws     = (float*)d_ws;
    float* msg    = ws;                                  // N*64
    float* deg    = msg + (size_t)N_NODES * F_IN;        // N
    float* gbuf   = deg + N_NODES;                       // S*128
    float* energy = gbuf + (size_t)N_SUB * HID;          // G
    float* normb  = energy + N_GRP;                      // G
    const size_t zero_n = (size_t)N_NODES * F_IN + N_NODES +
                          (size_t)N_SUB * HID + 2 * N_GRP;

    zero_kernel<<<4096, 256, 0, stream>>>(ws, zero_n);

    edge_scatter_kernel<<<N_EDGES / 256, 256, 0, stream>>>(
        x, edge_index, msg, deg);

    conv_pool_kernel<<<N_NODES / 128, 256, 0, stream>>>(
        msg, deg, x, W_l, W_r, b_conv, node_batch, gbuf);

    mlp_head_kernel<<<N_SUB / 128, 256, 0, stream>>>(
        gbuf, W1, b1, W2, b2, weights, subgraph_batch, energy, normb);

    finalize_kernel<<<1, 64, 0, stream>>>(energy, normb, out);
}
